// StructuralAttention_3994319585616
// MI455X (gfx1250) — matrix operand order repaired and checked by reading, not compiled, not run on hardware
//
#include <hip/hip_runtime.h>
#include <hip/hip_bf16.h>

#define D_MODEL   256
#define NUM_HEADS 8
#define HEAD_DIM  32
#define BATCH     2
#define SEQ       2048
#define SCALE     0.17677669529663687f   // 1/sqrt(32)
#define SCPITCH   2052                   // padded LDS pitch (floats): 2052%64=4 -> conflict-free tile stores

typedef __bf16 bf16_t;
typedef __attribute__((ext_vector_type(16))) bf16_t bf16x16;
typedef __attribute__((ext_vector_type(8)))  bf16_t bf16x8;
typedef __attribute__((ext_vector_type(8)))  float  f32x8;

__device__ __forceinline__ f32x8 wmma_bf16(bf16x16 a, bf16x16 b, f32x8 c) {
  // D = A(16x32 bf16) * B(32x16 bf16) + C(16x16 f32)
  return __builtin_amdgcn_wmma_f32_16x16x32_bf16(false, a, false, b, (short)0, c, false, false);
}

// ---- fragment loaders (wave32 layouts per CDNA5 ISA 7.12.2) -----------------
// A 16x32 bf16: lane l holds row m=l&15; elems 0..7 = K half*8+j, elems 8..15 = K 16+half*8+j
__device__ __forceinline__ bf16x16 load_a_f32(const float* p, int ld) {
  const int lane = threadIdx.x & 31, half = lane >> 4, m = lane & 15;
  const float* r = p + m * ld + half * 8;
  bf16x16 a;
#pragma unroll
  for (int j = 0; j < 8; ++j) { a[j] = (bf16_t)r[j]; a[8 + j] = (bf16_t)r[16 + j]; }
  return a;
}
__device__ __forceinline__ bf16x16 load_a_bf16(const bf16_t* p, int ld) {
  const int lane = threadIdx.x & 31, half = lane >> 4, m = lane & 15;
  const bf16_t* r = p + m * ld + half * 8;
  bf16x8 lo = *(const bf16x8*)(r);
  bf16x8 hi = *(const bf16x8*)(r + 16);
  bf16x16 a;
#pragma unroll
  for (int j = 0; j < 8; ++j) { a[j] = lo[j]; a[8 + j] = hi[j]; }
  return a;
}
// B 32x16 from a source S where B[k][n] = S[n][k] (rows of S are columns of B).
// lane l: col n=l&15; elems 0..7 = K half*8+j, elems 8..15 = K 16+half*8+j (from S row n).
__device__ __forceinline__ bf16x16 load_b_colsrc_bf16(const bf16_t* p, int ld) {
  const int lane = threadIdx.x & 31, half = lane >> 4, n = lane & 15;
  const bf16_t* r = p + n * ld + half * 8;
  bf16x8 lo = *(const bf16x8*)(r);
  bf16x8 hi = *(const bf16x8*)(r + 16);
  bf16x16 b;
#pragma unroll
  for (int j = 0; j < 8; ++j) { b[j] = lo[j]; b[8 + j] = hi[j]; }
  return b;
}
__device__ __forceinline__ bf16x16 load_b_colsrc_f32(const float* p, int ld) {
  const int lane = threadIdx.x & 31, half = lane >> 4, n = lane & 15;
  const float* r = p + n * ld + half * 8;
  bf16x16 b;
#pragma unroll
  for (int j = 0; j < 16; ++j) b[j] = (bf16_t)r[j + (j & 8)];
  return b;
}

// ---- kernel 1: QKV projections, x @ W.T + b, write bf16 (v transposed) ------
__global__ void __launch_bounds__(256) proj_qkv_kernel(
    const float* __restrict__ q_in, const float* __restrict__ k_in, const float* __restrict__ v_in,
    const float* __restrict__ Wq, const float* __restrict__ bq,
    const float* __restrict__ Wk, const float* __restrict__ bk,
    const float* __restrict__ Wv, const float* __restrict__ bv,
    bf16_t* __restrict__ q_s,   // [B,H,L,32]
    bf16_t* __restrict__ k_s,   // [B,H,L,32]
    bf16_t* __restrict__ vT_s)  // [B,H,32,L]
{
  const int wave = (blockIdx.x * blockDim.x + threadIdx.x) >> 5;
  const int mt  = wave & 255;          // 256 M tiles (B*L/16)
  const int nt  = (wave >> 8) & 15;    // 16 N tiles
  const int mat = wave >> 12;          // 0=q 1=k 2=v
  const float *x, *W, *bias;
  if (mat == 0)      { x = q_in; W = Wq; bias = bq; }
  else if (mat == 1) { x = k_in; W = Wk; bias = bk; }
  else               { x = v_in; W = Wv; bias = bv; }

  const int mb = mt * 16, nb = nt * 16;
  f32x8 acc = {};
#pragma unroll
  for (int kb = 0; kb < D_MODEL; kb += 32) {
    bf16x16 a = load_a_f32(x + (size_t)mb * D_MODEL + kb, D_MODEL);
    bf16x16 b = load_b_colsrc_f32(W + (size_t)nb * D_MODEL + kb, D_MODEL); // B[k][n]=W[nb+n][kb+k]
    acc = wmma_bf16(a, b, acc);
  }
  const int lane = threadIdx.x & 31, half = lane >> 4, ln = lane & 15;
  const int N = nb + ln;
  const float badd = bias[N];
  const int h = N >> 5, dh = N & 31;
#pragma unroll
  for (int r = 0; r < 8; ++r) {
    const int t  = mb + r + 8 * half;      // 0..4095 over B*L
    const int b_ = t >> 11, l = t & 2047;
    const float val = acc[r] + badd;
    if (mat == 0)
      q_s[((((size_t)b_ * NUM_HEADS + h) * SEQ + l) << 5) + dh] = (bf16_t)val;
    else if (mat == 1)
      k_s[((((size_t)b_ * NUM_HEADS + h) * SEQ + l) << 5) + dh] = (bf16_t)val;
    else
      vT_s[(((size_t)b_ * NUM_HEADS + h) * HEAD_DIM + dh) * SEQ + l] = (bf16_t)val;
  }
}

// ---- kernel 2: fused scores + bias + mask + softmax + attn write + context --
// one workgroup (8 waves) per (b, h, 16-query-row block); scores kept in LDS.
// bias/mask are read once (non-temporal), attn is written once (non-temporal),
// so the bf16 q/k/vT working set (~6 MB) stays resident in the 192 MB L2.
__global__ void __launch_bounds__(256) attn_kernel(
    const bf16_t* __restrict__ q_s, const bf16_t* __restrict__ k_s, const bf16_t* __restrict__ vT_s,
    const float* __restrict__ sbias, const float* __restrict__ amask,
    float* __restrict__ attn_out,   // [B,H,L,L]
    float* __restrict__ ctx_ws)     // [B*L, 256] fp32
{
  extern __shared__ float sc[];              // 16 x SCPITCH floats (131,328 B)
  const int wg = blockIdx.x;                 // 0..2047
  const int qt = wg & 127;                   // q tile (16 rows)
  const int h  = (wg >> 7) & 7;
  const int b  = wg >> 10;
  const int wave = threadIdx.x >> 5;
  const int lane = threadIdx.x & 31, half = lane >> 4, ln = lane & 15;

  const bf16_t* qp  = q_s  + (((size_t)b * NUM_HEADS + h) * SEQ + qt * 16) * HEAD_DIM;
  const bf16_t* kp  = k_s  + (((size_t)b * NUM_HEADS + h) * SEQ) * HEAD_DIM;
  const bf16_t* vTp = vT_s + (((size_t)b * NUM_HEADS + h) * HEAD_DIM) * SEQ;
  const size_t  row0 = ((size_t)(b * NUM_HEADS + h) * SEQ + qt * 16) * (size_t)SEQ;
  const float*  bp = sbias + row0;
  const float*  mp = amask + row0;
  float*        ap = attn_out + row0;

  // A fragment: q tile 16x32 (K = whole head dim), identical for all k tiles
  const bf16x16 afrag = load_a_bf16(qp, HEAD_DIM);

  // ---- phase 1: score tiles (each wave: 16 consecutive k-tiles = 256 cols)
  for (int kt = wave * 16; kt < wave * 16 + 16; ++kt) {
    const int kb = kt * 16;
    bf16x16 bfrag = load_b_colsrc_bf16(kp + (size_t)kb * HEAD_DIM, HEAD_DIM); // B[d][n]=k[kb+n][d]
    f32x8 c = {};
    c = wmma_bf16(afrag, bfrag, c);
#pragma unroll
    for (int r = 0; r < 8; ++r) {
      const int M = r + 8 * half;
      const int col = kb + ln;
      const float bm = __builtin_nontemporal_load(bp + (size_t)M * SEQ + col) +
                       __builtin_nontemporal_load(mp + (size_t)M * SEQ + col);
      sc[M * SCPITCH + col] = c[r] * SCALE + bm;
    }
  }
  __syncthreads();

  // ---- phase 2: softmax; wave w handles rows 2w, 2w+1; writes attn_weights
#pragma unroll
  for (int rr = 0; rr < 2; ++rr) {
    const int row = wave * 2 + rr;
    float* srow = sc + row * SCPITCH;
    float m = -1e30f;
    for (int c = lane; c < SEQ; c += 32) m = fmaxf(m, srow[c]);
#pragma unroll
    for (int off = 16; off; off >>= 1) m = fmaxf(m, __shfl_xor(m, off, 32));
    float sum = 0.f;
    for (int c = lane; c < SEQ; c += 32) { float e = __expf(srow[c] - m); srow[c] = e; sum += e; }
#pragma unroll
    for (int off = 16; off; off >>= 1) sum += __shfl_xor(sum, off, 32);
    const float inv = 1.f / sum;
    float* arow = ap + (size_t)row * SEQ;
    for (int c = lane; c < SEQ; c += 32) {
      float p = srow[c] * inv;
      srow[c] = p;
      __builtin_nontemporal_store(p, arow + c);   // stream: written exactly once
    }
  }
  __syncthreads();

  // ---- phase 3: context = P(16x2048) @ V(2048x32); K split across waves
  f32x8 acc0 = {}, acc1 = {};
#pragma unroll
  for (int i = 0; i < 8; ++i) {
    const int kb = (wave * 8 + i) * 32;
    // A from LDS fp32 -> bf16
    bf16x16 a;
    const float* r0 = sc + ln * SCPITCH + kb + half * 8;
#pragma unroll
    for (int j = 0; j < 8; ++j) { a[j] = (bf16_t)r0[j]; a[8 + j] = (bf16_t)r0[16 + j]; }
    // B from vT: B[k][n] = v[kb+k][d]; vT row d contiguous in k
    bf16x16 b0 = load_b_colsrc_bf16(vTp + kb, SEQ);
    bf16x16 b1 = load_b_colsrc_bf16(vTp + (size_t)16 * SEQ + kb, SEQ);
    acc0 = wmma_bf16(a, b0, acc0);
    acc1 = wmma_bf16(a, b1, acc1);
  }
  __syncthreads();
  // cross-wave reduction via LDS (reuse sc): wave region = 512 floats
  float* part = sc;
#pragma unroll
  for (int r = 0; r < 8; ++r) {
    part[wave * 512 +       r * 32 + lane] = acc0[r];
    part[wave * 512 + 256 + r * 32 + lane] = acc1[r];
  }
  __syncthreads();
  for (int idx = threadIdx.x; idx < 512; idx += 256) {
    float s = 0.f;
#pragma unroll
    for (int w = 0; w < 8; ++w) s += part[w * 512 + idx];
    const int tile = idx >> 8, r = (idx >> 5) & 7, l2 = idx & 31;
    const int M = r + 8 * (l2 >> 4);
    const int d = tile * 16 + (l2 & 15);
    const int t = qt * 16 + M;
    ctx_ws[((size_t)b * SEQ + t) * D_MODEL + h * HEAD_DIM + d] = s;
  }
}

// ---- kernel 3: output projection ctx @ Wo.T + bo -> fp32 output -------------
__global__ void __launch_bounds__(256) out_proj_kernel(
    const float* __restrict__ ctx, const float* __restrict__ Wo,
    const float* __restrict__ bo, float* __restrict__ out)
{
  const int wave = (blockIdx.x * blockDim.x + threadIdx.x) >> 5;
  const int mt = wave & 255, nt = wave >> 8;
  const int mb = mt * 16, nb = nt * 16;
  f32x8 acc = {};
#pragma unroll
  for (int kb = 0; kb < D_MODEL; kb += 32) {
    bf16x16 a = load_a_f32(ctx + (size_t)mb * D_MODEL + kb, D_MODEL);
    bf16x16 b = load_b_colsrc_f32(Wo + (size_t)nb * D_MODEL + kb, D_MODEL);
    acc = wmma_bf16(a, b, acc);
  }
  const int lane = threadIdx.x & 31, half = lane >> 4, ln = lane & 15;
  const float badd = bo[nb + ln];
#pragma unroll
  for (int r = 0; r < 8; ++r)
    __builtin_nontemporal_store(acc[r] + badd,
        out + (size_t)(mb + r + 8 * half) * D_MODEL + nb + ln);
}

extern "C" void kernel_launch(void* const* d_in, const int* in_sizes, int n_in,
                              void* d_out, int out_size, void* d_ws, size_t ws_size,
                              hipStream_t stream) {
  (void)in_sizes; (void)n_in; (void)out_size; (void)ws_size;
  const float* q_in  = (const float*)d_in[0];
  const float* k_in  = (const float*)d_in[1];
  const float* v_in  = (const float*)d_in[2];
  const float* sbias = (const float*)d_in[3];
  const float* amask = (const float*)d_in[4];
  const float* Wq = (const float*)d_in[5];  const float* bq = (const float*)d_in[6];
  const float* Wk = (const float*)d_in[7];  const float* bk = (const float*)d_in[8];
  const float* Wv = (const float*)d_in[9];  const float* bv = (const float*)d_in[10];
  const float* Wo = (const float*)d_in[11]; const float* bo = (const float*)d_in[12];

  float* out  = (float*)d_out;                                  // [B*L*256]
  float* attn = out + (size_t)BATCH * SEQ * D_MODEL;            // [B,H,L,L]

  char* ws = (char*)d_ws;
  const size_t nqkv = (size_t)BATCH * NUM_HEADS * SEQ * HEAD_DIM; // 1,048,576 elems
  bf16_t* q_s  = (bf16_t*)ws;
  bf16_t* k_s  = q_s + nqkv;
  bf16_t* vT_s = k_s + nqkv;
  float*  ctx  = (float*)(ws + 3 * nqkv * sizeof(bf16_t));       // 4 MB fp32, total ws use ~10 MB

  // 1) projections: 3 mats x 16 ntiles x 256 mtiles = 12288 waves -> 1536 blocks
  proj_qkv_kernel<<<1536, 256, 0, stream>>>(q_in, k_in, v_in, Wq, bq, Wk, bk, Wv, bv,
                                            q_s, k_s, vT_s);
  // 2) fused attention: B*H*(L/16) = 2048 workgroups, 131,328 B dynamic LDS
  attn_kernel<<<2048, 256, 16 * SCPITCH * sizeof(float), stream>>>(
      q_s, k_s, vT_s, sbias, amask, attn, ctx);
  // 3) output projection: 256 mtiles x 16 ntiles = 4096 waves -> 512 blocks
  out_proj_kernel<<<512, 256, 0, stream>>>(ctx, Wo, bo, out);
}
